// NHRepNet_23424751632495
// MI455X (gfx1250) — matrix operand order repaired and checked by reading, not compiled, not run on hardware
//
#include <hip/hip_runtime.h>
#include <hip/hip_bf16.h>

typedef _Float16 half_t;
typedef __attribute__((ext_vector_type(16))) _Float16 v16h;
typedef __attribute__((ext_vector_type(8)))  _Float16 v8h;
typedef __attribute__((ext_vector_type(8)))  float    v8f;

#define RS 264                 // LDS activation row stride in halves (256 + 8 pad)
#define MT 8                   // M-tiles (of 16 points) per workgroup
#define MPTS (MT * 16)         // 128 points per workgroup
#define NH_BETA 100.0f
#define INV_BETA 0.01f
#define INV_SQRT2 0.70710678118654752f

// workspace layout (in half_t units unless noted)
#define OFF_W0   0             // [256 n][32 k]   f16 * beta, k>=3 zero
#define OFF_W16  8192          // 6 x [256][256]  f16 * beta (layers 1..6; layer3 n>=253 zero)
#define OFF_W7   401408        // [16 n][256 k]   f16 (unscaled), n>=8 zero
#define OFF_BIAS 405504        // 8 x 256 f32 (zero-padded; layers 0..6 * beta), viewed as float*
#define PREP_N   407552        // 405504 half slots + 2048 bias float slots

union AU { v16h v; v8h h[2]; };

// ---------------------------------------------------------------------------
// Prep: f32 -> f16 transposed/padded weights (beta-scaled) + padded f32 biases
// ---------------------------------------------------------------------------
__global__ __launch_bounds__(256)
void nhrep_prep(const float* __restrict__ W0, const float* __restrict__ b0,
                const float* __restrict__ W1, const float* __restrict__ b1,
                const float* __restrict__ W2, const float* __restrict__ b2,
                const float* __restrict__ W3, const float* __restrict__ b3,
                const float* __restrict__ W4, const float* __restrict__ b4,
                const float* __restrict__ W5, const float* __restrict__ b5,
                const float* __restrict__ W6, const float* __restrict__ b6,
                const float* __restrict__ W7, const float* __restrict__ b7,
                half_t* __restrict__ ws) {
  int idx = blockIdx.x * 256 + threadIdx.x;
  if (idx >= PREP_N) return;

  if (idx < OFF_W16) {                       // W0: [3][256] -> [256][32], * beta
    int n = idx >> 5, k = idx & 31;
    ws[idx] = (half_t)((k < 3) ? NH_BETA * W0[k * 256 + n] : 0.0f);
  } else if (idx < OFF_W7) {                 // W1..W6: [K][N] -> [N][K], * beta
    int r = idx - OFF_W16;
    int l = r >> 16;                         // 0..5 -> layers 1..6
    int q = r & 65535;
    int n = q >> 8, k = q & 255;
    const float* Wsrc;
    switch (l) {
      case 0: Wsrc = W1; break;
      case 1: Wsrc = W2; break;
      case 2: Wsrc = W3; break;
      case 3: Wsrc = W4; break;
      case 4: Wsrc = W5; break;
      default: Wsrc = W6; break;
    }
    float v;
    if (l == 2) v = (n < 253) ? Wsrc[k * 253 + n] : 0.0f;   // layer3: 256x253
    else        v = Wsrc[k * 256 + n];
    ws[idx] = (half_t)(NH_BETA * v);
  } else if (idx < OFF_BIAS) {               // W7: [256][8] -> [16][256], unscaled
    int r = idx - OFF_W7;
    int n = r >> 8, k = r & 255;
    ws[idx] = (half_t)((n < 8) ? W7[k * 8 + n] : 0.0f);
  } else {                                   // biases, zero-padded; 0..6 * beta
    int r = idx - OFF_BIAS;
    int l = r >> 8, n = r & 255;
    const float* bs;
    int len;
    switch (l) {
      case 0: bs = b0; len = 256; break;
      case 1: bs = b1; len = 256; break;
      case 2: bs = b2; len = 256; break;
      case 3: bs = b3; len = 253; break;
      case 4: bs = b4; len = 256; break;
      case 5: bs = b5; len = 256; break;
      case 6: bs = b6; len = 256; break;
      default: bs = b7; len = 8;  break;
    }
    float scale = (l == 7) ? 1.0f : NH_BETA;
    float* bdst = (float*)(ws + OFF_BIAS);
    bdst[r] = (n < len) ? scale * bs[n] : 0.0f;
  }
}

// ---------------------------------------------------------------------------
// Fused MLP: 128 points/WG; 8 waves, each owns a 32-col N-slice over all 8
// M-tiles. B fragments loaded once per WG per layer; A fragments register
// double-buffered so ds_load latency hides under WMMA execution.
// ---------------------------------------------------------------------------
__global__ __launch_bounds__(256)
void nhrep_fused(const float* __restrict__ x,
                 const half_t* __restrict__ ws,
                 float* __restrict__ out,
                 int npts) {
  __shared__ half_t act[2][MPTS * RS];
  __shared__ float  bv[MPTS][8];

  const float* biasAll = (const float*)(ws + OFF_BIAS);
  const int p0   = blockIdx.x * MPTS;
  const int tid  = threadIdx.x;
  const int lane = tid & 31;
  const int wv   = tid >> 5;     // n-slice (0..7): columns [wv*32, wv*32+32)
  const int lm   = lane & 15;
  const int lg   = lane >> 4;
  const int n0   = wv * 32 + lm;       // column of n-tile 0
  const int n1   = n0 + 16;            // column of n-tile 1
  const int rowb = lm * RS;            // row offset base (add mt*16*RS)

  // ---- init layer-0 activation tile: [MPTS pts][32 k], k>=3 zero ----
  for (int idx = tid; idx < MPTS * 32; idx += 256) {
    int m = idx >> 5, k = idx & 31;
    float v = 0.0f;
    if (k < 3 && (p0 + m) < npts) v = x[(p0 + m) * 3 + k];
    act[0][m * RS + k] = (half_t)v;
  }
  __syncthreads();

  int buf = 0;

  v8f acc[MT][2];

  // ======================= layer 0: K = 32 (padded) =======================
  {
    const half_t* Wt = ws + OFF_W0;
    const float bn0 = biasAll[n0];
    const float bn1 = biasAll[n1];
#pragma unroll
    for (int mt = 0; mt < MT; ++mt)
#pragma unroll
      for (int i = 0; i < 8; ++i) { acc[mt][0][i] = bn0; acc[mt][1][i] = bn1; }

    const int kbB = 16 * lg;
    v16h b0 = *(const v16h*)&Wt[n0 * 32 + kbB];
    v16h b1 = *(const v16h*)&Wt[n1 * 32 + kbB];
    const half_t* A = &act[buf][0];
    const int kb = 16 * lg;

    AU a[2];   // register double-buffer for A fragments
    a[0].h[0] = *(const v8h*)&A[rowb + kb];
    a[0].h[1] = *(const v8h*)&A[rowb + kb + 8];
#pragma unroll
    for (int mt = 0; mt < MT; ++mt) {
      if (mt + 1 < MT) {
        a[(mt + 1) & 1].h[0] = *(const v8h*)&A[rowb + (mt + 1) * (16 * RS) + kb];
        a[(mt + 1) & 1].h[1] = *(const v8h*)&A[rowb + (mt + 1) * (16 * RS) + kb + 8];
      }
      acc[mt][0] = __builtin_amdgcn_wmma_f32_16x16x32_f16(
          false, a[mt & 1].v, false, b0, (short)0, acc[mt][0], false, false);
      acc[mt][1] = __builtin_amdgcn_wmma_f32_16x16x32_f16(
          false, a[mt & 1].v, false, b1, (short)0, acc[mt][1], false, false);
    }
    // epilogue: softplus(z)/beta
    half_t* O = &act[buf ^ 1][0];
#pragma unroll
    for (int nt = 0; nt < 2; ++nt) {
      int n = (nt == 0) ? n0 : n1;
#pragma unroll
      for (int mt = 0; mt < MT; ++mt)
#pragma unroll
        for (int r = 0; r < 8; ++r) {
          int m = mt * 16 + r + 8 * lg;
          float z = acc[mt][nt][r];
          float t = fmaxf(z, 0.0f) + __logf(1.0f + __expf(-__builtin_fabsf(z)));
          O[m * RS + n] = (half_t)(t * INV_BETA);
        }
    }
    buf ^= 1;
  }
  __syncthreads();

  // ======================= layers 1..6: K = 256 ===========================
  for (int layer = 1; layer < 7; ++layer) {
    const half_t* Wt = ws + OFF_W16 + (layer - 1) * 65536;
    const float*  bs = biasAll + layer * 256;
    const half_t* A  = &act[buf][0];
    half_t*       O  = &act[buf ^ 1][0];

    const float bn0 = bs[n0];
    const float bn1 = bs[n1];
#pragma unroll
    for (int mt = 0; mt < MT; ++mt)
#pragma unroll
      for (int i = 0; i < 8; ++i) { acc[mt][0][i] = bn0; acc[mt][1][i] = bn1; }

    const int kbBoff = 16 * lg;
    v16h b0 = *(const v16h*)&Wt[n0 * 256 + kbBoff];
    v16h b1 = *(const v16h*)&Wt[n1 * 256 + kbBoff];

    for (int kt = 0; kt < 8; ++kt) {
      v16h nb0, nb1;
      if (kt < 7) {  // pipeline next K-step's B fragments
        int kbBn = (kt + 1) * 32 + kbBoff;
        nb0 = *(const v16h*)&Wt[n0 * 256 + kbBn];
        nb1 = *(const v16h*)&Wt[n1 * 256 + kbBn];
      }
      const int kb = kt * 32 + 16 * lg;

      AU a[2];   // register double-buffer for A fragments
      a[0].h[0] = *(const v8h*)&A[rowb + kb];
      a[0].h[1] = *(const v8h*)&A[rowb + kb + 8];
#pragma unroll
      for (int mt = 0; mt < MT; ++mt) {
        if (mt + 1 < MT) {
          a[(mt + 1) & 1].h[0] = *(const v8h*)&A[rowb + (mt + 1) * (16 * RS) + kb];
          a[(mt + 1) & 1].h[1] = *(const v8h*)&A[rowb + (mt + 1) * (16 * RS) + kb + 8];
        }
        acc[mt][0] = __builtin_amdgcn_wmma_f32_16x16x32_f16(
            false, a[mt & 1].v, false, b0, (short)0, acc[mt][0], false, false);
        acc[mt][1] = __builtin_amdgcn_wmma_f32_16x16x32_f16(
            false, a[mt & 1].v, false, b1, (short)0, acc[mt][1], false, false);
      }
      b0 = nb0; b1 = nb1;
    }

    const bool  isSkip = (layer == 3);
    const float cscale = isSkip ? (INV_BETA * INV_SQRT2) : INV_BETA;
#pragma unroll
    for (int nt = 0; nt < 2; ++nt) {
      int n = (nt == 0) ? n0 : n1;
      bool keep = !isSkip || (n < 253);
#pragma unroll
      for (int mt = 0; mt < MT; ++mt)
#pragma unroll
        for (int r = 0; r < 8; ++r) {
          int m = mt * 16 + r + 8 * lg;
          float z = acc[mt][nt][r];
          float t = fmaxf(z, 0.0f) + __logf(1.0f + __expf(-__builtin_fabsf(z)));
          if (keep) O[m * RS + n] = (half_t)(t * cscale);
        }
    }
    if (isSkip) {  // skip concat: channels 253..255 = x / sqrt(2)
      for (int idx = tid; idx < 3 * MPTS; idx += 256) {
        int m = idx / 3, k = idx - 3 * m;
        float v = 0.0f;
        if ((p0 + m) < npts) v = x[(p0 + m) * 3 + k];
        O[m * RS + 253 + k] = (half_t)(v * INV_SQRT2);
      }
    }
    __syncthreads();
    buf ^= 1;
  }

  // ========== layer 7: 8 outputs (padded to 16); wave wv -> m-tile wv ======
  {
    const half_t* Wt = ws + OFF_W7;
    const float*  bs = biasAll + 7 * 256;
    const half_t* A  = &act[buf][0];
    const int arow = (wv * 16 + lm) * RS;

    v8f a7;
    float bn = bs[lm];   // zero beyond n=7
#pragma unroll
    for (int i = 0; i < 8; ++i) a7[i] = bn;

    AU a[2];
    v16h bfr[2];
    a[0].h[0] = *(const v8h*)&A[arow + 16 * lg];
    a[0].h[1] = *(const v8h*)&A[arow + 16 * lg + 8];
    bfr[0]    = *(const v16h*)&Wt[lm * 256 + 16 * lg];
#pragma unroll
    for (int kt = 0; kt < 8; ++kt) {
      if (kt + 1 < 8) {   // pipeline A and B across K-steps
        const int kbn = (kt + 1) * 32;
        a[(kt + 1) & 1].h[0] = *(const v8h*)&A[arow + kbn + 16 * lg];
        a[(kt + 1) & 1].h[1] = *(const v8h*)&A[arow + kbn + 16 * lg + 8];
        bfr[(kt + 1) & 1]    = *(const v16h*)&Wt[lm * 256 + kbn + 16 * lg];
      }
      a7 = __builtin_amdgcn_wmma_f32_16x16x32_f16(
          false, a[kt & 1].v, false, bfr[kt & 1], (short)0, a7, false, false);
    }
    if (lm < 8) {
#pragma unroll
      for (int r = 0; r < 8; ++r) bv[wv * 16 + r + 8 * lg][lm] = a7[r];
    }
  }
  __syncthreads();

  // ---- CSG tree + output: max(v0, v1, min(v2,v3), min(v4, v5, max(v6,v7))) ----
  for (int m = tid; m < MPTS; m += 256) {
    if ((p0 + m) >= npts) continue;
    float v0 = bv[m][0], v1 = bv[m][1], v2 = bv[m][2], v3 = bv[m][3];
    float v4 = bv[m][4], v5 = bv[m][5], v6 = bv[m][6], v7 = bv[m][7];
    float i2 = fminf(fminf(v4, v5), fmaxf(v6, v7));
    float i1 = fminf(v2, v3);
    float h  = fmaxf(fmaxf(v0, v1), fmaxf(i1, i2));
    float* o = out + (size_t)(p0 + m) * 9;
    o[0] = h;
    o[1] = v0; o[2] = v1; o[3] = v2; o[4] = v3;
    o[5] = v4; o[6] = v5; o[7] = v6; o[8] = v7;
  }
}

// ---------------------------------------------------------------------------
extern "C" void kernel_launch(void* const* d_in, const int* in_sizes, int n_in,
                              void* d_out, int out_size, void* d_ws, size_t ws_size,
                              hipStream_t stream) {
  const float* x = (const float*)d_in[0];
  const float* W[8];
  const float* b[8];
  for (int l = 0; l < 8; ++l) {
    W[l] = (const float*)d_in[1 + 2 * l];
    b[l] = (const float*)d_in[2 + 2 * l];
  }
  half_t* ws  = (half_t*)d_ws;
  float*  out = (float*)d_out;

  nhrep_prep<<<(PREP_N + 255) / 256, 256, 0, stream>>>(
      W[0], b[0], W[1], b[1], W[2], b[2], W[3], b[3],
      W[4], b[4], W[5], b[5], W[6], b[6], W[7], b[7], ws);

  int npts = in_sizes[0] / 3;                 // 100000
  int nblk = (npts + MPTS - 1) / MPTS;        // 782
  nhrep_fused<<<nblk, 256, 0, stream>>>(x, ws, out, npts);
}
